// ObjectAttentionBlock_65893388255896
// MI455X (gfx1250) — matrix operand order repaired and checked by reading, not compiled, not run on hardware
//
#include <hip/hip_runtime.h>

// ---------------------------------------------------------------------------
// Types
// ---------------------------------------------------------------------------
typedef __attribute__((ext_vector_type(16))) __bf16 v16bf;
typedef __attribute__((ext_vector_type(8)))  __bf16 v8bf;
typedef __attribute__((ext_vector_type(8)))  float  v8f;

__device__ inline __bf16 to_bf16(float x)  { return (__bf16)x; }
__device__ inline __bf16 to_bf16(__bf16 x) { return x; }

template <typename T> __device__ inline T from_f32(float v);
template <> __device__ inline float  from_f32<float>(float v)  { return v; }
template <> __device__ inline __bf16 from_f32<__bf16>(float v) { return (__bf16)v; }

// ---------------------------------------------------------------------------
// Big fused GEMM:  out[n,o,p] = relu( scale[o] * sum_c W[o,c]*X[n,c,p] + bias[o] )
// W: [O, CIN] f32 row-major.  X: [N, CIN, PP].  out: [N, O, PP].
// Block: 256 threads (8 waves). Tile BM=128, BN=128, BK=32, double-buffered LDS.
// Waves: 2 (M) x 4 (N); each wave computes 64x32 = 4x2 WMMA 16x16 tiles.
// CIN, PP are compile-time so all hot-loop addressing is 32-bit shift/add.
// ---------------------------------------------------------------------------
template <typename TIn, typename TOut, int CIN, int PP>
__global__ __launch_bounds__(256)
void gemm_bn_relu(const float* __restrict__ W,
                  const TIn*  __restrict__ X,
                  const float* __restrict__ scale,
                  const float* __restrict__ bias,
                  TOut* __restrict__ out,
                  int O)
{
    constexpr int BM = 128, BN = 128, BK = 32;
    constexpr int LDA = 40;          // bf16 elems per sA row (32 + pad) -> 80B pitch
    constexpr int LDB = 40;          // bf16 elems per sB "column" (K-major per pixel)
    constexpr int NT  = CIN / BK;    // K-steps (16 or 8)

    __shared__ __align__(16) __bf16 sA[2][BM * LDA];   // [buf][row][k]
    __shared__ __align__(16) __bf16 sB[2][BN * LDB];   // [buf][pixel][k]
    __shared__ float sS[BM];
    __shared__ float sBi[BM];

    const int tid   = threadIdx.x;
    const int lane  = tid & 31;
    const int wave  = tid >> 5;
    const int waveM = wave >> 2;      // 0..1
    const int waveN = wave & 3;       // 0..3
    const int lane16   = lane & 15;
    const int laneHalf = lane >> 4;   // 0/1

    const int n     = blockIdx.z;
    const int oBase = blockIdx.y * BM;
    const int pBase = blockIdx.x * BN;

    const TIn* __restrict__ Xb = X   + (size_t)n * CIN * PP;
    TOut*      __restrict__ Ob = out + (size_t)n * O   * PP;

    if (tid < BM) { sS[tid] = scale[oBase + tid]; sBi[tid] = bias[oBase + tid]; }

    v8f acc[4][2] = {};

    // Per-thread global-load assignments
    const int aKchunk  = tid & 7;     // 8 chunks of 4 floats along K
    const int aRowBase = tid >> 3;    // 0..31; 4 passes cover 128 rows
    const int bP  = tid & 127;        // pixel within tile
    const int bKq = tid >> 7;         // 0/1 -> k = bKq*16 + i

    // Register staging for the next tile (enables batched, unwaited loads)
    float4 aReg[4];
    TIn    bReg[16];

    auto loadTile = [&](int it) {
        const int k0 = it * BK;
        #pragma unroll
        for (int rr = 0; rr < 4; ++rr) {
            const int row = aRowBase + rr * 32;
            aReg[rr] = *(const float4*)(W + (oBase + row) * CIN + k0 + aKchunk * 4);
        }
        #pragma unroll
        for (int i = 0; i < 16; ++i)
            bReg[i] = Xb[(k0 + bKq * 16 + i) * PP + pBase + bP];
        // prefetch one tile further ahead
        if (it + 1 < NT) {
            __builtin_prefetch(W  + (oBase + aRowBase) * CIN + (it + 1) * BK, 0, 3);
            __builtin_prefetch(Xb + ((it + 1) * BK + bKq * 16) * PP + pBase + bP, 0, 3);
        }
    };
    auto storeTile = [&](int buf) {
        #pragma unroll
        for (int rr = 0; rr < 4; ++rr) {
            const int row = aRowBase + rr * 32;
            __bf16* dst = &sA[buf][row * LDA + aKchunk * 4];
            dst[0] = to_bf16(aReg[rr].x); dst[1] = to_bf16(aReg[rr].y);
            dst[2] = to_bf16(aReg[rr].z); dst[3] = to_bf16(aReg[rr].w);
        }
        __bf16* dst = &sB[buf][bP * LDB + bKq * 16];
        #pragma unroll
        for (int i = 0; i < 16; ++i) dst[i] = to_bf16(bReg[i]);
    };

    // ---- pipeline prologue
    loadTile(0);
    storeTile(0);
    __syncthreads();

    // ---- main loop: read LDS[cur], preload next tile into regs, commit to LDS[1-cur]
    for (int it = 0; it < NT; ++it) {
        const int cur = it & 1;
        if (it + 1 < NT) loadTile(it + 1);   // global loads in flight during WMMA

        // fragments per documented VGPR layouts
        v16bf aF[4];
        #pragma unroll
        for (int mt = 0; mt < 4; ++mt) {
            const int row = waveM * 64 + mt * 16 + lane16;
            // elements 0..7 : K =      laneHalf*8 + 0..7
            // elements 8..15: K = 16 + laneHalf*8 + 0..7
            const v8bf lo = *(const v8bf*)&sA[cur][row * LDA + laneHalf * 8];
            const v8bf hi = *(const v8bf*)&sA[cur][row * LDA + 16 + laneHalf * 8];
            aF[mt] = __builtin_shufflevector(lo, hi, 0,1,2,3,4,5,6,7,
                                                     8,9,10,11,12,13,14,15);
        }
        v16bf bF[2];
        #pragma unroll
        for (int nt = 0; nt < 2; ++nt) {
            const int col = waveN * 32 + nt * 16 + lane16;
            // elements 0..7 : K = laneHalf*8 + 0..7; elements 8..15: K = 16 + laneHalf*8 + 0..7
            const v8bf lo = *(const v8bf*)&sB[cur][col * LDB + laneHalf * 8];
            const v8bf hi = *(const v8bf*)&sB[cur][col * LDB + 16 + laneHalf * 8];
            bF[nt] = __builtin_shufflevector(lo, hi, 0,1,2,3,4,5,6,7,
                                                     8,9,10,11,12,13,14,15);
        }
        #pragma unroll
        for (int mt = 0; mt < 4; ++mt)
            #pragma unroll
            for (int nt = 0; nt < 2; ++nt)
                acc[mt][nt] = __builtin_amdgcn_wmma_f32_16x16x32_bf16(
                    false, aF[mt], false, bF[nt], (short)0, acc[mt][nt],
                    false, false);

        if (it + 1 < NT) storeTile(1 - cur); // other buffer: no hazard with readers
        __syncthreads();
    }

    // ---- epilogue: BN + ReLU, scatter per D-matrix layout
    #pragma unroll
    for (int mt = 0; mt < 4; ++mt) {
        #pragma unroll
        for (int nt = 0; nt < 2; ++nt) {
            const int col = pBase + waveN * 32 + nt * 16 + lane16;
            #pragma unroll
            for (int e = 0; e < 8; ++e) {
                const int rloc = waveM * 64 + mt * 16 + laneHalf * 8 + e;
                float v = acc[mt][nt][e] * sS[rloc] + sBi[rloc];
                v = v > 0.0f ? v : 0.0f;
                Ob[(oBase + rloc) * PP + col] = from_f32<TOut>(v);
            }
        }
    }
}

// ---------------------------------------------------------------------------
// Tiny projection for the proxy path (M=19):  out[n,o,m] = relu(s*W·x + b)
// ---------------------------------------------------------------------------
__global__ void small_proj(const float* __restrict__ W,   // [O, Cin]
                           const float* __restrict__ X,   // [N, Cin, M]
                           const float* __restrict__ s,
                           const float* __restrict__ b,
                           float* __restrict__ out,       // [N, O, M]
                           int O, int Cin, int M, int N)
{
    const int idx = blockIdx.x * blockDim.x + threadIdx.x;
    if (idx >= N * O * M) return;
    const int m = idx % M;
    const int o = (idx / M) % O;
    const int n = idx / (M * O);
    const float* x = X + (size_t)n * Cin * M + m;
    const float* w = W + (size_t)o * Cin;
    float acc = 0.0f;
    for (int c = 0; c < Cin; ++c) acc = fmaf(w[c], x[(size_t)c * M], acc);
    float v = acc * s[o] + b[o];
    out[(size_t)n * O * M + (size_t)o * M + m] = v > 0.0f ? v : 0.0f;
}

// ---------------------------------------------------------------------------
// Attention over object regions (M=19): one thread per pixel.
// sim[m] = scl * sum_c q[c,p]*key[c,m]; softmax over m; ctx[c,p]=sum_m a[m]*v[c,m]
// ---------------------------------------------------------------------------
__global__ __launch_bounds__(256)
void attention_kernel(const __bf16* __restrict__ q,    // [N, Kc, P] bf16
                      const float*  __restrict__ key,  // [N, Kc, M] f32
                      const float*  __restrict__ val,  // [N, Kc, M] f32
                      __bf16* __restrict__ ctx,        // [N, Kc, P] bf16
                      int Kc, int P, float scl)
{
    constexpr int M  = 19;
    constexpr int MP = 20;   // padded row
    __shared__ float sK[256 * MP];
    __shared__ float sV[256 * MP];

    const int n = blockIdx.y;
    const int p = blockIdx.x * 256 + threadIdx.x;

    for (int i = threadIdx.x; i < Kc * M; i += 256) {
        const int c = i / M, m = i % M;
        sK[c * MP + m] = key[(size_t)n * Kc * M + i];
        sV[c * MP + m] = val[(size_t)n * Kc * M + i];
    }
    __syncthreads();

    const __bf16* qb = q + (size_t)n * Kc * P + p;
    float sim[M];
    #pragma unroll
    for (int m = 0; m < M; ++m) sim[m] = 0.0f;

    for (int c = 0; c < Kc; ++c) {
        const float qc = (float)qb[(size_t)c * P];
        const float* kr = &sK[c * MP];
        #pragma unroll
        for (int m = 0; m < M; ++m) sim[m] = fmaf(qc, kr[m], sim[m]);
    }

    float mx = -3.4e38f;
    #pragma unroll
    for (int m = 0; m < M; ++m) { sim[m] *= scl; mx = sim[m] > mx ? sim[m] : mx; }
    float sum = 0.0f;
    #pragma unroll
    for (int m = 0; m < M; ++m) { sim[m] = __expf(sim[m] - mx); sum += sim[m]; }
    const float inv = 1.0f / sum;
    #pragma unroll
    for (int m = 0; m < M; ++m) sim[m] *= inv;

    __bf16* cb = ctx + (size_t)n * Kc * P + p;
    for (int c = 0; c < Kc; ++c) {
        const float* vr = &sV[c * MP];
        float a = 0.0f;
        #pragma unroll
        for (int m = 0; m < M; ++m) a = fmaf(sim[m], vr[m], a);
        cb[(size_t)c * P] = (__bf16)a;
    }
}

// ---------------------------------------------------------------------------
// Host launcher
// ---------------------------------------------------------------------------
extern "C" void kernel_launch(void* const* d_in, const int* in_sizes, int n_in,
                              void* d_out, int out_size, void* d_ws, size_t ws_size,
                              hipStream_t stream) {
    (void)in_sizes; (void)n_in; (void)out_size; (void)ws_size;
    constexpr int N = 8, C = 512, Kc = 256, H = 128, Wd = 128, M = 19;
    constexpr int P = H * Wd;   // 16384

    const float* feat  = (const float*)d_in[0];
    const float* proxy = (const float*)d_in[1];
    const float* wp1 = (const float*)d_in[2];
    const float* sp1 = (const float*)d_in[3];
    const float* bp1 = (const float*)d_in[4];
    const float* wp2 = (const float*)d_in[5];
    const float* sp2 = (const float*)d_in[6];
    const float* bp2 = (const float*)d_in[7];
    const float* wo1 = (const float*)d_in[8];
    const float* so1 = (const float*)d_in[9];
    const float* bo1 = (const float*)d_in[10];
    const float* wo2 = (const float*)d_in[11];
    const float* so2 = (const float*)d_in[12];
    const float* bo2 = (const float*)d_in[13];
    const float* wdw = (const float*)d_in[14];
    const float* sd  = (const float*)d_in[15];
    const float* bd  = (const float*)d_in[16];
    const float* wu  = (const float*)d_in[17];
    const float* su  = (const float*)d_in[18];
    const float* bu  = (const float*)d_in[19];
    float* outp = (float*)d_out;

    // ---- workspace carving
    char* base = (char*)d_ws;
    auto align256 = [](size_t x) { return (x + 255) & ~(size_t)255; };
    const size_t smallSz = align256((size_t)N * Kc * M * sizeof(float));
    const size_t bigSz   = align256((size_t)N * Kc * P * sizeof(__bf16));
    float*  tsmall = (float*)(base);
    float*  keyB   = (float*)(base + smallSz);
    float*  valB   = (float*)(base + 2 * smallSz);
    __bf16* bufA   = (__bf16*)(base + 3 * smallSz);            // t1, then ctx
    __bf16* bufQ   = (__bf16*)(base + 3 * smallSz + bigSz);    // q

    // ---- proxy path (tiny)
    {
        const int tot = N * Kc * M;
        small_proj<<<(tot + 255) / 256, 256, 0, stream>>>(wo1, proxy, so1, bo1,
                                                          tsmall, Kc, C, M, N);
        small_proj<<<(tot + 255) / 256, 256, 0, stream>>>(wo2, tsmall, so2, bo2,
                                                          keyB, Kc, Kc, M, N);
        small_proj<<<(tot + 255) / 256, 256, 0, stream>>>(wdw, proxy, sd, bd,
                                                          valB, Kc, C, M, N);
    }

    // ---- pixel path: t1 = proj(feat, wp1) ; q = proj(t1, wp2)
    {
        dim3 grid(P / 128, Kc / 128, N);
        gemm_bn_relu<float, __bf16, C, P><<<grid, 256, 0, stream>>>(
            wp1, feat, sp1, bp1, bufA, Kc);
        gemm_bn_relu<__bf16, __bf16, Kc, P><<<grid, 256, 0, stream>>>(
            wp2, bufA, sp2, bp2, bufQ, Kc);
    }

    // ---- attention: ctx (reuses bufA)
    {
        dim3 grid(P / 256, N);
        const float scl = 1.0f / 16.0f;   // Kc^-0.5, Kc=256
        attention_kernel<<<grid, 256, 0, stream>>>(bufQ, keyB, valB, bufA,
                                                   Kc, P, scl);
    }

    // ---- up-projection to output
    {
        dim3 grid(P / 128, C / 128, N);
        gemm_bn_relu<__bf16, float, Kc, P><<<grid, 256, 0, stream>>>(
            wu, bufA, su, bu, outp, C);
    }
}
